// GerbilizerRNNConv_42829413876222
// MI455X (gfx1250) — matrix operand order repaired and checked by reading, not compiled, not run on hardware
//
#include <hip/hip_runtime.h>
#include <cstdint>

typedef __attribute__((ext_vector_type(16))) _Float16 v16h;
typedef __attribute__((ext_vector_type(4)))  _Float16 v4h;
typedef __attribute__((ext_vector_type(8)))  float    v8f;

#define DEVI static __device__ __forceinline__

// ---- problem dims ----
constexpr int Bn = 16, Tn = 64, Cn = 4, Hn = 129, Wn = 2, HID = 32;
constexpr int Nn = Hn * Wn;                 // 258 columns (n = 2*h + w)
constexpr int NT = 17;                      // 17 n-tiles of 16 (272, tail padded)
constexpr int JOBS = 2 * NT;                // (mgroup in {0,1}) x ntile
constexpr int GUARD = 2;                    // halo columns for kh shift
constexpr int COLS = 276;                   // 2 + 272 + 2
constexpr int ROWP = 64;                    // panel rows (both layers, padded)
constexpr int K0IN = 36, K1IN = 64;

// ---- weight fragment packing (wave32 WMMA A layout) ----
constexpr int FRAGS = 96;                   // 48 per layer: (mtile8 x kh3 x kt2)
constexpr int FRAG_HALVES = 512;            // 32 lanes * 16 halves = 1 KB

// ---- global scratch layout (float units; A-pack occupies first A_FLOATS) ----
constexpr int A_FLOATS = FRAGS * FRAG_HALVES / 2;   // 24576
constexpr int CSZ  = 264 * 32;              // c-state per sample per layer, [n][hid]
constexpr int HLSZ = HID * Nn;              // h_last per sample, [hid][n]
constexpr int C0_F = A_FLOATS;
constexpr int C1_F = C0_F + Bn * CSZ;
constexpr int HL_F = C1_F + Bn * CSZ;
constexpr int XH_F = HL_F + Bn * HLSZ;      // x as f16, [b][t][n][c], c fastest
constexpr int XH_HALVES = Bn * Tn * Nn * Cn;

// ---- dynamic LDS layout (halves) ----
constexpr int P0_H = 2 * COLS * ROWP;       // ping-pong layer0 input panel
constexpr int P1_H = 2 * COLS * ROWP;       // ping-pong layer1 input panel
constexpr int SMEM_BYTES = (P0_H + P1_H) * 2 + (264 + 256) * 4;  // 143,392 B

DEVI float sig_(float x) { return 1.0f / (1.0f + __expf(-x)); }
DEVI float tanh_(float x) {
  x = fminf(15.0f, fmaxf(-15.0f, x));
  float e = __expf(2.0f * x);
  return (e - 1.0f) / (e + 1.0f);
}

// Pack conv weights into WMMA A fragments (f16), one block per fragment.
// Fragment f: layer = f/48; fl = f%48 = mtile*6 + kh*2 + kt.
// A element [M=l16][K=kt*32+klocal], oc = mtile*16 + l16, K = ic (panel row).
// K >= K_in is zero-padded, which also nullifies panel pad rows read by B.
__global__ void prep_pack_kernel(const float* __restrict__ w0,
                                 const float* __restrict__ w1,
                                 _Float16* __restrict__ apack) {
  int f = blockIdx.x;           // 0..95
  int lane = threadIdx.x;       // 0..31
  bool isA1 = f >= 48;
  int fl = isA1 ? f - 48 : f;
  int mtile = fl / 6, kh = (fl % 6) / 2, kt = fl & 1;
  int l16 = lane & 15, hi = lane >> 4;
  int oc = mtile * 16 + l16;
  int KIN = isA1 ? K1IN : K0IN;
  const float* w = isA1 ? w1 : w0;
  v16h v;
#pragma unroll
  for (int s = 0; s < 16; ++s) {
    // lanes 0-15: K in {0..15}; lanes 16-31: K in {16..31}
    int klocal = (hi ? 16 : 0) + s;
    int k = kt * 32 + klocal;
    float val = (k < KIN) ? w[(oc * KIN + k) * 3 + kh] : 0.0f;
    v[s] = (_Float16)val;
  }
  *(v16h*)(apack + (size_t)f * FRAG_HALVES + lane * 16) = v;
}

// Convert x to f16 in [b][t][n][c] order (c fastest: 4 halves = 8 B per column,
// async-DMA-able straight into panel rows 0..3).
__global__ void prep_x_kernel(const float* __restrict__ x,
                              _Float16* __restrict__ xh) {
  const int total = Bn * Tn * Nn * Cn;
  for (int i = blockIdx.x * blockDim.x + threadIdx.x; i < total;
       i += gridDim.x * blockDim.x) {
    int c = i & 3;
    int rem = i >> 2;
    int n = rem % Nn;
    int bt = rem / Nn;
    xh[i] = (_Float16)x[((size_t)bt * Cn + c) * Nn + n];
  }
}

// One ConvLSTM layer step for this workgroup (GEMM via WMMA + cell update).
// pin: input panel [col][row] f16 (col = n + GUARD); rows = input channels.
// `wave` must be an SGPR value (readfirstlane'd) so the job loop is scalar and
// EXEC stays all-ones around the WMMAs.
DEVI void layer_step(const _Float16* __restrict__ pin,
                     _Float16* pwA, int offA,
                     _Float16* pwB, int offB,
                     const _Float16* __restrict__ apack,
                     const v8f* __restrict__ bias4,   // per-wave i,f,o,g bias
                     float* __restrict__ cbuf,        // [n*32 + hid]
                     float* __restrict__ hlast, bool doLast,
                     int lane, int wave) {
  const int l16 = lane & 15, hi = lane >> 4;
  const int mg = wave & 1;                 // jid = wave + 8k -> mg wave-invariant
  for (int jid = wave; jid < JOBS; jid += 8) {
    const int nc = (jid >> 1) * 16 + l16;  // output column n
    v8f a0 = {}, a1 = {}, a2 = {}, a3 = {};  // i, f, o, g accumulators
#pragma unroll
    for (int kh = 0; kh < 3; ++kh) {
      const int col = nc + 2 * kh;         // n + 2*(kh-1) + GUARD
#pragma unroll
      for (int kt = 0; kt < 2; ++kt) {
        // B frag: lanes 0-15 hold K rows k0..k0+15, lanes 16-31 k0+16..k0+31
        v16h bf = *(const v16h*)(pin + col * ROWP + kt * 32 + (hi ? 16 : 0));
        // gate gi uses mtile = gi*2+mg -> frag = gi*12 + (mg*3+kh)*2 + kt
        const _Float16* ab =
            apack + ((size_t)((mg * 3 + kh) * 2 + kt)) * FRAG_HALVES + lane * 16;
        v16h af;
        af = *(const v16h*)(ab + 0 * 12 * FRAG_HALVES);
        a0 = __builtin_amdgcn_wmma_f32_16x16x32_f16(false, af, false, bf,
                                                    (short)0, a0, false, false);
        af = *(const v16h*)(ab + 1 * 12 * FRAG_HALVES);
        a1 = __builtin_amdgcn_wmma_f32_16x16x32_f16(false, af, false, bf,
                                                    (short)0, a1, false, false);
        af = *(const v16h*)(ab + 2 * 12 * FRAG_HALVES);
        a2 = __builtin_amdgcn_wmma_f32_16x16x32_f16(false, af, false, bf,
                                                    (short)0, a2, false, false);
        af = *(const v16h*)(ab + 3 * 12 * FRAG_HALVES);
        a3 = __builtin_amdgcn_wmma_f32_16x16x32_f16(false, af, false, bf,
                                                    (short)0, a3, false, false);
      }
    }
    // LSTM cell update; C/D layout: lane holds column nc, VGPR r -> hid0+r.
    const int hid0 = mg * 16 + (hi ? 8 : 0);
    const bool valid = nc < Nn;
    float* cptr = cbuf + nc * 32 + hid0;
    v8f cold = {};
    if (valid) cold = *(const v8f*)cptr;
    alignas(16) _Float16 hh[8];
    float hf[8];
#pragma unroll
    for (int r = 0; r < 8; ++r) {
      float iv = a0[r] + bias4[0][r];
      float fv = a1[r] + bias4[1][r];
      float ov = a2[r] + bias4[2][r];
      float gv = a3[r] + bias4[3][r];
      float cn = sig_(fv) * cold[r] + sig_(iv) * tanh_(gv);
      float hv = sig_(ov) * tanh_(cn);
      cold[r] = cn;
      hh[r] = (_Float16)hv;
      hf[r] = hv;
    }
    if (valid) {
      *(v8f*)cptr = cold;
      const int colw = nc + GUARD;
      _Float16* dA = pwA + colw * ROWP + offA + hid0;
      *(v4h*)dA = *(const v4h*)hh;
      *(v4h*)(dA + 4) = *(const v4h*)(hh + 4);
      if (pwB) {
        _Float16* dB = pwB + colw * ROWP + offB + hid0;
        *(v4h*)dB = *(const v4h*)hh;
        *(v4h*)(dB + 4) = *(const v4h*)(hh + 4);
      }
      if (doLast) {
#pragma unroll
        for (int r = 0; r < 8; ++r) hlast[(hid0 + r) * Nn + nc] = hf[r];
      }
    }
  }
}

__global__ void __launch_bounds__(256, 1)
convlstm_kernel(const _Float16* __restrict__ xh, const int* __restrict__ lens,
                const float* __restrict__ b0c, const float* __restrict__ b1c,
                const float* __restrict__ ninw, const float* __restrict__ ninb,
                const float* __restrict__ fc1w, const float* __restrict__ fc1b,
                const float* __restrict__ fc2w, const float* __restrict__ fc2b,
                const _Float16* __restrict__ apack,
                float* __restrict__ wsf, float* __restrict__ out) {
  extern __shared__ char smem_raw[];
  _Float16* in0T = (_Float16*)smem_raw;       // 2 x [COLS][ROWP] f16
  _Float16* in1T = in0T + P0_H;               // 2 x [COLS][ROWP] f16
  float* flat = (float*)(in1T + P1_H);        // 264 f32 (head)
  float* f1o = flat + 264;                    // 256 f32 (head)

  const int b = blockIdx.x;
  const int tid = threadIdx.x;
  const int lane = tid & 31;
  // wave index is identical across the 32 lanes; force it into an SGPR so the
  // job loop is a scalar loop and EXEC is provably all-ones at every WMMA.
  const int wave = __builtin_amdgcn_readfirstlane(tid >> 5);

  float* c0 = wsf + C0_F + b * CSZ;
  float* c1 = wsf + C1_F + b * CSZ;
  float* hl = wsf + HL_F + b * HLSZ;

  // zero c-state (scratch is poisoned) and panels (incl. guard cols + pad rows)
  for (int i = tid; i < CSZ; i += 256) { c0[i] = 0.f; c1[i] = 0.f; }
  for (int i = tid; i < P0_H + P1_H; i += 256) in0T[i] = (_Float16)0.f;
  __threadfence();
  __syncthreads();

  // per-wave gate biases (mg and hid0 are wave-invariant)
  const int hid0w = (wave & 1) * 16 + ((lane >> 4) ? 8 : 0);
  v8f bi0[4], bi1[4];
#pragma unroll
  for (int gi = 0; gi < 4; ++gi) {
    bi0[gi] = *(const v8f*)(b0c + gi * 32 + hid0w);
    bi1[gi] = *(const v8f*)(b1c + gi * 32 + hid0w);
  }

  const int len = lens[b];
  for (int t = 0; t < Tn; ++t) {
    const int cur = t & 1, nxt = cur ^ 1;
    _Float16* p0c = in0T + cur * (COLS * ROWP);
    _Float16* p0n = in0T + nxt * (COLS * ROWP);
    _Float16* p1c = in1T + cur * (COLS * ROWP);
    _Float16* p1n = in1T + nxt * (COLS * ROWP);

    // async-DMA x_t straight into layer0 panel rows 0..3 (8 B per column)
    const _Float16* xsrc = xh + ((size_t)(b * Tn + t) * Nn) * 4;
    for (int i = tid; i < Nn; i += 256) {
      uint32_t ldso =
          (uint32_t)((const char*)(p0c + (i + GUARD) * ROWP) - smem_raw);
      uint64_t ga = (uint64_t)(uintptr_t)(xsrc + (size_t)i * 4);
      asm volatile("global_load_async_to_lds_b64 %0, %1, off"
                   :: "v"(ldso), "v"(ga) : "memory");
    }
    asm volatile("s_wait_asynccnt 0x0" ::: "memory");
    __syncthreads();
    // layer0: reads [x_t; h0(t-1)] from p0c; h0(t) -> p0n rows 4.., p1c rows 0..
    layer_step(p0c, p0n, Cn, p1c, 0, apack, bi0, c0,
               (float*)nullptr, false, lane, wave);
    __syncthreads();
    // layer1: reads [h0(t); h1(t-1)] from p1c; h1(t) -> p1n rows 32..
    layer_step(p1c, p1n, HID, (_Float16*)nullptr, 0,
               apack + 48 * FRAG_HALVES, bi1, c1, hl, (t == len - 1),
               lane, wave);
  }
  __threadfence();
  __syncthreads();

  // head: 1x1 channel reduce -> flat(258) -> fc1(256) -> fc2(2)
  for (int n = tid; n < Nn; n += 256) {
    float s = ninb[0];
    for (int c = 0; c < HID; ++c) s += hl[c * Nn + n] * ninw[c];
    flat[n] = s;
  }
  __syncthreads();
  {
    float s = fc1b[tid];
    const float* wr = fc1w + tid * Nn;
    for (int n = 0; n < Nn; ++n) s += flat[n] * wr[n];
    f1o[tid] = s;
  }
  __syncthreads();
  if (tid < 2) {
    float s = fc2b[tid];
    const float* wr = fc2w + tid * 256;
    for (int k = 0; k < 256; ++k) s += f1o[k] * wr[k];
    out[b * 2 + tid] = s;
  }
}

extern "C" void kernel_launch(void* const* d_in, const int* in_sizes, int n_in,
                              void* d_out, int out_size, void* d_ws,
                              size_t ws_size, hipStream_t stream) {
  (void)in_sizes; (void)n_in; (void)out_size; (void)ws_size;
  const float* x    = (const float*)d_in[0];
  const int*   lens = (const int*)d_in[1];
  const float* w0   = (const float*)d_in[2];
  const float* b0   = (const float*)d_in[3];
  const float* w1   = (const float*)d_in[4];
  const float* b1   = (const float*)d_in[5];
  const float* ninw = (const float*)d_in[6];
  const float* ninb = (const float*)d_in[7];
  const float* fc1w = (const float*)d_in[8];
  const float* fc1b = (const float*)d_in[9];
  const float* fc2w = (const float*)d_in[10];
  const float* fc2b = (const float*)d_in[11];
  float* out = (float*)d_out;
  float* wsf = (float*)d_ws;
  _Float16* apack = (_Float16*)d_ws;
  _Float16* xh = (_Float16*)(wsf + XH_F);

  (void)hipFuncSetAttribute((const void*)convlstm_kernel,
                            hipFuncAttributeMaxDynamicSharedMemorySize,
                            SMEM_BYTES);

  prep_pack_kernel<<<FRAGS, 32, 0, stream>>>(w0, w1, apack);
  prep_x_kernel<<<512, 256, 0, stream>>>(x, xh);
  convlstm_kernel<<<Bn, 256, SMEM_BYTES, stream>>>(
      xh, lens, b0, b1, ninw, ninb, fc1w, fc1b, fc2w, fc2b, apack, wsf, out);
}
